// AttentionPool_72816875536608
// MI455X (gfx1250) — matrix operand order repaired and checked by reading, not compiled, not run on hardware
//
#include <hip/hip_runtime.h>
#include <hip/hip_bf16.h>
#include <stdint.h>

// ---------------- types for CDNA5 WMMA ----------------
typedef __attribute__((ext_vector_type(16))) __bf16 v16bf;
typedef __attribute__((ext_vector_type(8)))  float  v8f;

union BFrag {
    v16bf    v;
    uint32_t u[8];
};

#define HID   512          // hidden size (K and N of the GEMM)
#define MTILE 128          // rows per workgroup (8 waves x 16)
#define KCH   16           // K chunks of 32 (16*32 = 512)
#define NSUB  32           // N subtiles of 16 (32*16 = 512)
#define BPITCH 520         // LDS pitch (halves): 260 dwords % 64 banks = 4 -> conflict-free frag reads

// round-to-nearest-even fp32 -> bf16 (as raw ushort)
__device__ __forceinline__ unsigned short f2bf(float f) {
    uint32_t u = __float_as_uint(f);
    u += 0x7FFFu + ((u >> 16) & 1u);
    return (unsigned short)(u >> 16);
}

// ---------------- kernel 0: W1 (fp32 [k][c]) -> W1t (bf16 [c][k]) ----------------
__global__ __launch_bounds__(256) void conv_w1_kernel(const float* __restrict__ W1,
                                                      unsigned short* __restrict__ W1t) {
    int i = blockIdx.x * 256 + threadIdx.x;   // 0 .. 512*512-1
    int c = i >> 9;
    int k = i & (HID - 1);
    W1t[(c << 9) | k] = f2bf(W1[(k << 9) | c]);
}

// ---------------- kernel 1: graph boundaries via binary search ----------------
__global__ void starts_kernel(const int* __restrict__ bv, int* __restrict__ starts,
                              int n, int num_graphs) {
    int g = blockIdx.x * blockDim.x + threadIdx.x;
    if (g > num_graphs) return;
    int lo = 0, hi = n;
    while (lo < hi) {
        int mid = (lo + hi) >> 1;
        if (bv[mid] < g) lo = mid + 1; else hi = mid;
    }
    starts[g] = lo;   // lower_bound; starts[num_graphs] == n
}

// ---------------- kernel 2: gate = gelu(h @ W1 + b1) @ W2 + b2, fused ----------------
// __launch_bounds__(256, 2): cap VGPRs so >=2 waves/SIMD stay resident (hides the
// per-WMMA LDS waits) and avoid the >256-VGPR high-addressing mode (s_set_vgpr_msb).
__global__ __launch_bounds__(256, 2) void gate_kernel(const float* __restrict__ h,
                                                      const unsigned short* __restrict__ W1t,
                                                      const float* __restrict__ b1,
                                                      const float* __restrict__ W2,
                                                      const float* __restrict__ b2,
                                                      float* __restrict__ gate,
                                                      int N) {
    // double-buffered B tile: 16 cols x 512 K (bf16), padded pitch
    __shared__ __attribute__((aligned(16))) unsigned short sB[2][16 * BPITCH];

    const int tid  = threadIdx.x;
    const int wid  = tid >> 5;          // wave 0..7 -> m-subtile
    const int lane = tid & 31;
    const int col  = lane & 15;         // local column inside 16-wide subtile / M index
    const int grp  = lane >> 4;         // half-wave group
    const int rowBase = blockIdx.x * MTILE;

    // ---- register-cache this wave's A slab: 16 rows x K=512 as bf16 fragments ----
    // ISA A layout (16-bit 16x32): lane m=lane%16; VGPR r<4 -> K=2r, r>=4 -> K=8+2r; +8 for lanes 16..31
    BFrag af[KCH];
    {
        const long long row = (long long)rowBase + wid * 16 + col;
        const bool valid = row < (long long)N;
        const float* hrow = h + (size_t)(valid ? row : 0) * HID;
#pragma unroll
        for (int kc = 0; kc < KCH; ++kc) {
#pragma unroll
            for (int r = 0; r < 8; ++r) {
                const int pb = 2 * r + ((r >= 4) ? 8 : 0) + 8 * grp;
                float x0 = 0.f, x1 = 0.f;
                if (valid) {
                    const float2 xy = *(const float2*)(hrow + kc * 32 + pb);
                    x0 = xy.x; x1 = xy.y;
                }
                af[kc].u[r] = (uint32_t)f2bf(x0) | ((uint32_t)f2bf(x1) << 16);
            }
        }
    }

    // ---- cooperative stage of B subtile nt into buffer b ----
    // The 16 columns of W1t for subtile nt are one contiguous 16KB block:
    // batch 4x global_load_b128 into registers, then 4x ds_store_b128.
    auto stageB = [&](int nt, int b) {
        const uint4* src = (const uint4*)W1t + (((size_t)nt) << 10) + tid;  // nt*1024 uint4
        const uint4 t0 = src[0];
        const uint4 t1 = src[256];
        const uint4 t2 = src[512];
        const uint4 t3 = src[768];
        const int i0 = tid, i1 = tid + 256, i2 = tid + 512, i3 = tid + 768;
        *(uint4*)(&sB[b][(i0 >> 6) * BPITCH] + ((i0 & 63) << 3)) = t0;
        *(uint4*)(&sB[b][(i1 >> 6) * BPITCH] + ((i1 & 63) << 3)) = t1;
        *(uint4*)(&sB[b][(i2 >> 6) * BPITCH] + ((i2 & 63) << 3)) = t2;
        *(uint4*)(&sB[b][(i3 >> 6) * BPITCH] + ((i3 & 63) << 3)) = t3;
    };

    float gp[8];
#pragma unroll
    for (int r = 0; r < 8; ++r) gp[r] = 0.f;

    stageB(0, 0);

    for (int nt = 0; nt < NSUB; ++nt) {
        const int cur = nt & 1;
        __syncthreads();                    // buf[cur] ready; prior reads of buf[cur^1] done
        if (nt + 1 < NSUB) stageB(nt + 1, cur ^ 1);

        const uint32_t* bcol = (const uint32_t*)(&sB[cur][col * BPITCH]);

        v8f c = {0.f, 0.f, 0.f, 0.f, 0.f, 0.f, 0.f, 0.f};
#pragma unroll
        for (int kc = 0; kc < KCH; ++kc) {
            // B layout matches A: col=lane%16; VGPR r<4 -> K=2r, r>=4 -> K=8+2r; +8 for lanes 16..31
            BFrag bf;
#pragma unroll
            for (int r = 0; r < 8; ++r) {
                const int k = kc * 32 + 2 * r + ((r >= 4) ? 8 : 0) + 8 * grp;
                bf.u[r] = bcol[k >> 1];
            }
            c = __builtin_amdgcn_wmma_f32_16x16x32_bf16(
                    /*neg_a=*/false, af[kc].v, /*neg_b=*/false, bf.v,
                    /*c_mod=*/(short)0, c, /*reuse_a=*/false, /*reuse_b=*/false);
        }

        // fused epilogue: + b1, exact gelu, * W2, accumulate per-row gate partial
        const int gcol = nt * 16 + col;
        const float bias = b1[gcol];
        const float w2v  = W2[gcol];
#pragma unroll
        for (int r = 0; r < 8; ++r) {
            const float x  = c[r] + bias;
            const float gl = 0.5f * x * (1.0f + erff(x * 0.7071067811865475f));
            gp[r] = fmaf(w2v, gl, gp[r]);
        }
    }

    // reduce gate partials across the 16 lanes of each half-wave; row = r + 8*grp
    const float b2v = b2[0];
#pragma unroll
    for (int r = 0; r < 8; ++r) {
        float v = gp[r];
        v += __shfl_xor(v, 1);
        v += __shfl_xor(v, 2);
        v += __shfl_xor(v, 4);
        v += __shfl_xor(v, 8);
        if (col == 0) {
            const int row = rowBase + wid * 16 + r + 8 * grp;
            if (row < N) gate[row] = v + b2v;
        }
    }
}

// ---------------- kernel 3: per-graph segment softmax + weighted pooling ----------------
__global__ __launch_bounds__(256) void pool_kernel(const float* __restrict__ h,
                                                   const float* __restrict__ gate,
                                                   const int* __restrict__ starts,
                                                   float* __restrict__ out) {
    const int g   = blockIdx.x;
    const int tid = threadIdx.x;
    const int s = starts[g];
    const int e = starts[g + 1];
    float2* outp = (float2*)(out + (size_t)g * HID);

    if (s >= e) {                       // empty segment -> zeros (matches segment_sum)
        outp[tid] = make_float2(0.f, 0.f);
        return;
    }

    __shared__ float red[256];
    __shared__ float alpha[256];

    // segment max
    float m = -INFINITY;
    for (int i = s + tid; i < e; i += 256) m = fmaxf(m, gate[i]);
    red[tid] = m; __syncthreads();
    for (int off = 128; off > 0; off >>= 1) {
        if (tid < off) red[tid] = fmaxf(red[tid], red[tid + off]);
        __syncthreads();
    }
    m = red[0]; __syncthreads();

    // segment sum of exp
    float sm = 0.f;
    for (int i = s + tid; i < e; i += 256) sm += expf(gate[i] - m);
    red[tid] = sm; __syncthreads();
    for (int off = 128; off > 0; off >>= 1) {
        if (tid < off) red[tid] += red[tid + off];
        __syncthreads();
    }
    const float inv = 1.0f / red[0];
    __syncthreads();

    // weighted sum: thread owns 2 columns; alpha staged in LDS (broadcast reads)
    float a0 = 0.f, a1 = 0.f;
    for (int base = s; base < e; base += 256) {
        const int cnt = min(256, e - base);
        if (tid < cnt) alpha[tid] = expf(gate[base + tid] - m) * inv;
        __syncthreads();
        for (int i = 0; i < cnt; ++i) {
            const float2 hv = *(const float2*)(h + (size_t)(base + i) * HID + 2 * tid);
            const float al = alpha[i];
            a0 = fmaf(al, hv.x, a0);
            a1 = fmaf(al, hv.y, a1);
        }
        __syncthreads();
    }
    outp[tid] = make_float2(a0, a1);
}

// ---------------- host launcher ----------------
extern "C" void kernel_launch(void* const* d_in, const int* in_sizes, int n_in,
                              void* d_out, int out_size, void* d_ws, size_t ws_size,
                              hipStream_t stream) {
    const float* h   = (const float*)d_in[0];
    const int*   bv  = (const int*)  d_in[1];
    const float* W1  = (const float*)d_in[2];
    const float* b1  = (const float*)d_in[3];
    const float* W2  = (const float*)d_in[4];
    const float* b2  = (const float*)d_in[5];
    float* out = (float*)d_out;

    const int N  = in_sizes[0] / HID;
    const int NG = out_size / HID;

    // workspace layout: [W1t bf16 512x512][gate fp32 N][starts int NG+1]
    char* ws = (char*)d_ws;
    unsigned short* W1t = (unsigned short*)ws;
    size_t off = (size_t)HID * HID * sizeof(unsigned short);
    float* gate = (float*)(ws + off);
    off += ((size_t)N * sizeof(float) + 255) & ~(size_t)255;
    int* starts = (int*)(ws + off);

    conv_w1_kernel<<<(HID * HID) / 256, 256, 0, stream>>>(W1, W1t);
    starts_kernel<<<(NG + 1 + 255) / 256, 256, 0, stream>>>(bv, starts, N, NG);

    const int nwg = (N + MTILE - 1) / MTILE;
    gate_kernel<<<nwg, 256, 0, stream>>>(h, W1t, b1, W2, b2, gate, N);

    pool_kernel<<<NG, 256, 0, stream>>>(h, gate, starts, out);
}
